// LightweightMambaDecoder_89129161327127
// MI455X (gfx1250) — matrix operand order repaired and checked by reading, not compiled, not run on hardware
//
#include <hip/hip_runtime.h>
#include <hip/hip_bf16.h>

typedef __attribute__((ext_vector_type(16))) _Float16 v16h;
typedef __attribute__((ext_vector_type(8)))  _Float16 v8h;
typedef __attribute__((ext_vector_type(8)))  float    v8f;
typedef __attribute__((ext_vector_type(4)))  unsigned int uint32x4;
typedef __attribute__((ext_vector_type(8)))  int          int32x8;
typedef __attribute__((ext_vector_type(4)))  int          int32x4;

#define BATCH   16
#define SEQ     512
#define DMODEL  512
#define DINNER  1024
#define DSTATE  16
#define DTRANK  32
#define NLAYER  4
#define OUTDIM  256
#define TOKENS  (BATCH * SEQ)   // 8192

#define LDS_A 40   // halves per row: 32 data + 8 pad (=16B pad per 64B row)
#define LDS_B 40

__device__ __forceinline__ float fast_sigmoid(float x) {
    return 1.0f / (1.0f + __expf(-x));
}

// ---------------------------------------------------------------------------
// TDM: DMA one f16 tile (32 k-elems wide x `rows` tall, row stride
// `stride_elems`) from global into LDS at byte offset `lds_off`, adding
// 4 dwords (16B) of padding after every 16 dwords (64B = 32 halves) so the
// LDS layout has a 40-half row pitch.
// ---------------------------------------------------------------------------
__device__ __forceinline__ void tdm_load_tile_f16(const _Float16* gptr,
                                                  unsigned lds_off,
                                                  int stride_elems, int rows)
{
    unsigned long long ga = (unsigned long long)(uintptr_t)gptr;
    uint32x4 g0;
    g0[0] = 1u;                                     // count=1, user descriptor
    g0[1] = lds_off;                                // lds_addr (bytes)
    g0[2] = (unsigned)(ga & 0xFFFFFFFFu);           // global_addr[31:0]
    g0[3] = (unsigned)((ga >> 32) & 0x01FFFFFFu)    // global_addr[56:32]
          | (2u << 30);                             // type = 2 ("image")
    int32x8 g1;
    g1[0] = (1 << 16)      // data_size = 2 bytes
          | (1 << 20)      // pad_enable
          | (3 << 22)      // pad_interval: 16 dwords
          | (3 << 25);     // pad_amount:   4 dwords
    g1[1] = (32 << 16);                  // tensor_dim0[15:0]=32 (atomic_barrier=0)
    g1[2] = (rows << 16);                // tensor_dim0[31:16]=0, tensor_dim1[15:0]=rows
    g1[3] = (32 << 16);                  // tensor_dim1[31:16]=0, tile_dim0=32
    g1[4] = rows;                        // tile_dim1=rows, tile_dim2=0
    g1[5] = stride_elems;                // tensor_dim0_stride[31:0]
    g1[6] = 0;                           // stride[47:32]=0, tensor_dim1_stride lo=0
    g1[7] = 0;
    int32x4 z4 = {0, 0, 0, 0};
#if __clang_major__ >= 23
    int32x8 z8 = {0, 0, 0, 0, 0, 0, 0, 0};
    __builtin_amdgcn_tensor_load_to_lds(g0, g1, z4, z4, z8, 0);
#else
    __builtin_amdgcn_tensor_load_to_lds(g0, g1, z4, z4, 0);
#endif
}

// ---------------------------------------------------------------------------
// WMMA GEMM with TDM-staged, double-buffered LDS tiles.
//   C = epilogue( A[M,K](f16) * W[N,K]^T(f16) )
// Block tile 128(M) x 64(N), 8 waves (4x2), wave = 2x2 of v_wmma 16x16x32 f16.
// MODE: 0 plain, 1 +bias, 2 softplus(+bias).  WF32 -> float out, WF16 -> f16 out.
// ---------------------------------------------------------------------------
template<int MODE, bool WF32, bool WF16>
__global__ __launch_bounds__(256)
void gemm_tdm(const _Float16* __restrict__ A, const _Float16* __restrict__ W,
              const float* __restrict__ bias, float* __restrict__ Cf,
              _Float16* __restrict__ Ch, int K, int lda, int ldw, int ldc)
{
    __shared__ __align__(16) _Float16 smem[2 * 128 * LDS_A + 2 * 64 * LDS_B];

    const int tid   = threadIdx.x;
    const int lane  = tid & 31;
    const int wave  = tid >> 5;
    const int waveM = wave >> 1;      // 0..3
    const int waveN = wave & 1;       // 0..1
    const int bM    = blockIdx.y * 128;
    const int bN    = blockIdx.x * 64;
    const int lm    = lane & 15;
    const int kh    = lane >> 4;

    const unsigned smem_base = (unsigned)(uintptr_t)smem;
    const unsigned bsA       = 2u * 128 * LDS_A;   // halves offset of B-tile area

    v8f acc[2][2];
    #pragma unroll
    for (int s = 0; s < 2; ++s)
        #pragma unroll
        for (int t = 0; t < 2; ++t) {
            v8f z = {0.f,0.f,0.f,0.f,0.f,0.f,0.f,0.f};
            acc[s][t] = z;
        }

    const int nIter = K >> 5;
    if (wave == 0) {
        tdm_load_tile_f16(A + (size_t)bM * lda, smem_base, lda, 128);
        tdm_load_tile_f16(W + (size_t)bN * ldw, smem_base + bsA * 2u, ldw, 64);
    }

    for (int i = 0; i < nIter; ++i) {
        const int cur = i & 1;
        if (wave == 0) __builtin_amdgcn_s_wait_tensorcnt(0);
        __syncthreads();   // tiles in buf[cur] visible to all waves

        // prefetch next K-tile into the other buffer while we compute
        if (wave == 0 && (i + 1) < nIter) {
            const int kk  = (i + 1) * 32;
            const int nxt = cur ^ 1;
            tdm_load_tile_f16(A + (size_t)bM * lda + kk,
                              smem_base + (unsigned)(nxt * 128 * LDS_A) * 2u, lda, 128);
            tdm_load_tile_f16(W + (size_t)bN * ldw + kk,
                              smem_base + (bsA + (unsigned)(nxt * 64 * LDS_B)) * 2u, ldw, 64);
        }

        const _Float16* As = smem + cur * 128 * LDS_A;
        const _Float16* Bs = smem + bsA + cur * 64 * LDS_B;

        // fragments per CDNA5 16-bit A/B VGPR layouts
        v16h afrag[2], bfrag[2];
        #pragma unroll
        for (int s = 0; s < 2; ++s) {
            const int row = waveM * 32 + s * 16 + lm;
            v8h lo = *(const v8h*)(As + row * LDS_A + kh * 8);
            v8h hi = *(const v8h*)(As + row * LDS_A + kh * 8 + 16);
            afrag[s] = __builtin_shufflevector(lo, hi,
                0,1,2,3,4,5,6,7,8,9,10,11,12,13,14,15);
        }
        #pragma unroll
        for (int t = 0; t < 2; ++t) {
            const int row = waveN * 32 + t * 16 + lm;
            v8h lo = *(const v8h*)(Bs + row * LDS_B + kh * 8);
            v8h hi = *(const v8h*)(Bs + row * LDS_B + kh * 8 + 16);
            bfrag[t] = __builtin_shufflevector(lo, hi,
                0,1,2,3,4,5,6,7,8,9,10,11,12,13,14,15);
        }

        #pragma unroll
        for (int s = 0; s < 2; ++s)
            #pragma unroll
            for (int t = 0; t < 2; ++t)
                acc[s][t] = __builtin_amdgcn_wmma_f32_16x16x32_f16(
                    false, afrag[s], false, bfrag[t],
                    (short)0, acc[s][t], false, false);
        __syncthreads();   // everyone done reading buf[cur]
    }

    // epilogue (C/D layout: VGPR r -> m = r + 8*kh, n = lane&15)
    #pragma unroll
    for (int s = 0; s < 2; ++s) {
        #pragma unroll
        for (int t = 0; t < 2; ++t) {
            const int col = bN + waveN * 32 + t * 16 + lm;
            float bv = 0.0f;
            if (MODE != 0) bv = bias[col];
            #pragma unroll
            for (int r = 0; r < 8; ++r) {
                const int row = bM + waveM * 32 + s * 16 + r + 8 * kh;
                float v = acc[s][t][r];
                if (MODE == 1) v += bv;
                if (MODE == 2) {
                    v += bv;
                    v = (v > 20.0f) ? v : __logf(1.0f + __expf(v));  // softplus
                }
                if (WF32) Cf[(size_t)row * ldc + col] = v;
                if (WF16) Ch[(size_t)row * ldc + col] = (_Float16)v;
            }
        }
    }
}

// ---------------------------------------------------------------------------
__global__ __launch_bounds__(256)
void f32_to_f16_kernel(const float* __restrict__ src, _Float16* __restrict__ dst, int n)
{
    const int i = blockIdx.x * 256 + threadIdx.x;
    if (i < n) dst[i] = (_Float16)src[i];
}

// latent (16x512) @ W_seq^T + b_seq -> xseq (16x512) f32
__global__ __launch_bounds__(256)
void seqproj_kernel(const float* __restrict__ latent, const float* __restrict__ W_seq,
                    const float* __restrict__ b_seq, float* __restrict__ xseq)
{
    const int idx = blockIdx.x * 256 + threadIdx.x;     // 0..8191
    const int b = idx >> 9;
    const int d = idx & 511;
    const float* lrow = latent + (size_t)b * DMODEL;
    const float* wrow = W_seq + (size_t)d * DMODEL;
    float acc = b_seq[d];
    for (int k = 0; k < DMODEL; ++k) acc += lrow[k] * wrow[k];
    xseq[idx] = acc;
}

// x_h[(b*SEQ+l)*DMODEL + d] = (f16) xseq[b*DMODEL + d]
__global__ __launch_bounds__(256)
void broadcast_kernel(const float* __restrict__ xseq, _Float16* __restrict__ xh)
{
    const int i = blockIdx.x * 256 + threadIdx.x;       // < TOKENS*DMODEL = 2^22
    xh[i] = (_Float16)xseq[((i >> 18) << 9) | (i & 511)];
}

// depthwise causal conv (width 4) + bias + SiLU; writes f32 (scan) and f16 (GEMM)
__global__ __launch_bounds__(256)
void conv_silu_kernel(const float* __restrict__ xz, const float* __restrict__ convW,
                      const float* __restrict__ convB,
                      float* __restrict__ ucf, _Float16* __restrict__ uch)
{
    const int idx = blockIdx.x * 256 + threadIdx.x;     // < TOKENS*DINNER = 2^23
    const int c = idx & (DINNER - 1);
    const int t = idx >> 10;
    const int l = t & (SEQ - 1);
    float acc = convB[c];
    #pragma unroll
    for (int j = 0; j < 4; ++j) {
        const int l2 = l + j - 3;
        if (l2 >= 0)
            acc += convW[c * 4 + j] * xz[(size_t)(t + j - 3) * (2 * DINNER) + c];
    }
    const float v = acc * fast_sigmoid(acc);   // SiLU
    ucf[idx] = v;
    uch[idx] = (_Float16)v;
}

// selective scan; fuses D-skip + silu(z) gate; writes y as f16 (next GEMM input)
__global__ __launch_bounds__(256)
void scan_kernel(const float* __restrict__ dt,   const float* __restrict__ uc,
                 const float* __restrict__ xdbl, const float* __restrict__ xz,
                 const float* __restrict__ A_log, const float* __restrict__ D_skip,
                 _Float16* __restrict__ yh)
{
    __shared__ float Bs[32][DSTATE];
    __shared__ float Cs[32][DSTATE];

    const int b = blockIdx.y;
    const int c = blockIdx.x * 256 + threadIdx.x;

    float Av[DSTATE];
    #pragma unroll
    for (int n = 0; n < DSTATE; ++n)
        Av[n] = -__expf(A_log[(size_t)c * DSTATE + n]);
    const float Dv = D_skip[c];

    float h[DSTATE];
    #pragma unroll
    for (int n = 0; n < DSTATE; ++n) h[n] = 0.0f;

    for (int chunk = 0; chunk < SEQ / 32; ++chunk) {
        __syncthreads();
        #pragma unroll
        for (int i = 0; i < 4; ++i) {
            const int f = threadIdx.x + i * 256;   // 0..1023
            const int which = f >> 9;              // 0 = B, 1 = C
            const int g  = f & 511;
            const int li = g >> 4;
            const int n  = g & 15;
            const int t  = b * SEQ + chunk * 32 + li;
            const float v = xdbl[(size_t)t * 64 + DTRANK + which * DSTATE + n];
            if (which == 0) Bs[li][n] = v; else Cs[li][n] = v;
        }
        __syncthreads();

        for (int li = 0; li < 32; ++li) {
            const size_t t = (size_t)(b * SEQ + chunk * 32 + li);
            const float dt_t = dt[t * DINNER + c];
            const float u_t  = uc[t * DINNER + c];
            float yacc = 0.0f;
            #pragma unroll
            for (int n = 0; n < DSTATE; ++n) {
                const float dA = __expf(dt_t * Av[n]);
                h[n] = h[n] * dA + (dt_t * Bs[li][n]) * u_t;
                yacc += h[n] * Cs[li][n];
            }
            const float yv = yacc + u_t * Dv;
            const float z  = xz[t * (2 * DINNER) + DINNER + c];
            yh[t * DINNER + c] = (_Float16)(yv * (z * fast_sigmoid(z)));
        }
    }
}

// ---------------------------------------------------------------------------
extern "C" void kernel_launch(void* const* d_in, const int* in_sizes, int n_in,
                              void* d_out, int out_size, void* d_ws, size_t ws_size,
                              hipStream_t stream)
{
    (void)in_sizes; (void)n_in; (void)out_size; (void)ws_size;
    const float* latent  = (const float*)d_in[0];   // (16,512)
    const float* W_seq   = (const float*)d_in[1];   // (512,512)
    const float* b_seq   = (const float*)d_in[2];   // (512)
    const float* in_W    = (const float*)d_in[3];   // (4,2048,512)
    const float* conv_W  = (const float*)d_in[4];   // (4,1024,1,4)
    const float* conv_b  = (const float*)d_in[5];   // (4,1024)
    const float* xp_W    = (const float*)d_in[6];   // (4,64,1024)
    const float* dt_W    = (const float*)d_in[7];   // (4,1024,32)
    const float* dt_b    = (const float*)d_in[8];   // (4,1024)
    const float* A_log   = (const float*)d_in[9];   // (4,1024,16)
    const float* D_skip  = (const float*)d_in[10];  // (4,1024)
    const float* out_W   = (const float*)d_in[11];  // (4,512,1024)
    const float* W_out   = (const float*)d_in[12];  // (256,512)
    const float* b_out   = (const float*)d_in[13];  // (256)
    float* out = (float*)d_out;                     // (16,512,256)

    // ---- workspace layout (byte offsets; all regions 256B-aligned)
    char* w = (char*)d_ws;
    size_t o = 0;
    float* xseq = (float*)(w + o);       o += (size_t)BATCH * DMODEL * 4;        // 32 KB
    float* xz   = (float*)(w + o);       o += (size_t)TOKENS * 2 * DINNER * 4;   // 64 MB
    float* ucf  = (float*)(w + o);       o += (size_t)TOKENS * DINNER * 4;       // 32 MB
    float* xdf  = (float*)(w + o);       o += (size_t)TOKENS * 64 * 4;           //  2 MB
    float* dtf  = (float*)(w + o);       o += (size_t)TOKENS * DINNER * 4;       // 32 MB
    _Float16* xh   = (_Float16*)(w + o); o += (size_t)TOKENS * DMODEL * 2;       //  8 MB
    _Float16* uch  = (_Float16*)(w + o); o += (size_t)TOKENS * DINNER * 2;       // 16 MB
    _Float16* xdh  = (_Float16*)(w + o); o += (size_t)TOKENS * 64 * 2;           //  1 MB
    _Float16* yh   = (_Float16*)(w + o); o += (size_t)TOKENS * DINNER * 2;       // 16 MB
    _Float16* inWh  = (_Float16*)(w + o); o += (size_t)NLAYER * 2 * DINNER * DMODEL * 2;
    _Float16* xpWh  = (_Float16*)(w + o); o += (size_t)NLAYER * 64 * DINNER * 2;
    _Float16* dtWh  = (_Float16*)(w + o); o += (size_t)NLAYER * DINNER * DTRANK * 2;
    _Float16* outWh = (_Float16*)(w + o); o += (size_t)NLAYER * DMODEL * DINNER * 2;
    _Float16* WoutH = (_Float16*)(w + o); o += (size_t)OUTDIM * DMODEL * 2;

    // ---- one-time weight conversion f32 -> f16
    {
        const int n1 = NLAYER * 2 * DINNER * DMODEL;   // 4,194,304
        const int n2 = NLAYER * 64 * DINNER;           //   262,144
        const int n3 = NLAYER * DINNER * DTRANK;       //   131,072
        const int n4 = NLAYER * DMODEL * DINNER;       // 2,097,152
        const int n5 = OUTDIM * DMODEL;                //   131,072
        f32_to_f16_kernel<<<(n1 + 255) / 256, 256, 0, stream>>>(in_W,  inWh,  n1);
        f32_to_f16_kernel<<<(n2 + 255) / 256, 256, 0, stream>>>(xp_W,  xpWh,  n2);
        f32_to_f16_kernel<<<(n3 + 255) / 256, 256, 0, stream>>>(dt_W,  dtWh,  n3);
        f32_to_f16_kernel<<<(n4 + 255) / 256, 256, 0, stream>>>(out_W, outWh, n4);
        f32_to_f16_kernel<<<(n5 + 255) / 256, 256, 0, stream>>>(W_out, WoutH, n5);
    }

    // x = broadcast(latent @ W_seq^T + b_seq) over SEQ, stored f16
    seqproj_kernel<<<(BATCH * DMODEL) / 256, 256, 0, stream>>>(latent, W_seq, b_seq, xseq);
    broadcast_kernel<<<(TOKENS * DMODEL) / 256, 256, 0, stream>>>(xseq, xh);

    for (int l = 0; l < NLAYER; ++l) {
        const _Float16* inW_l  = inWh  + (size_t)l * 2 * DINNER * DMODEL;
        const float* convW_l   = conv_W + (size_t)l * DINNER * 4;
        const float* convb_l   = conv_b + (size_t)l * DINNER;
        const _Float16* xpW_l  = xpWh  + (size_t)l * 64 * DINNER;
        const _Float16* dtW_l  = dtWh  + (size_t)l * DINNER * DTRANK;
        const float* dtb_l     = dt_b  + (size_t)l * DINNER;
        const float* Alog_l    = A_log + (size_t)l * DINNER * DSTATE;
        const float* D_l       = D_skip + (size_t)l * DINNER;
        const _Float16* outW_l = outWh + (size_t)l * DMODEL * DINNER;

        // xz = x @ in_W^T   (M=8192, N=2048, K=512) -> f32
        gemm_tdm<0, true, false><<<dim3(2 * DINNER / 64, TOKENS / 128), 256, 0, stream>>>(
            xh, inW_l, nullptr, xz, nullptr, DMODEL, DMODEL, DMODEL, 2 * DINNER);

        // uc = silu(depthwise_conv(u) + b)  -> f32 + f16
        conv_silu_kernel<<<(TOKENS * DINNER) / 256, 256, 0, stream>>>(
            xz, convW_l, convb_l, ucf, uch);

        // x_dbl = uc @ xp_W^T   (M=8192, N=64, K=1024) -> f32 + f16
        gemm_tdm<0, true, true><<<dim3(64 / 64, TOKENS / 128), 256, 0, stream>>>(
            uch, xpW_l, nullptr, xdf, xdh, DINNER, DINNER, DINNER, 64);

        // dt = softplus(x_dbl[:, :32] @ dt_W^T + dt_b)  (M=8192, N=1024, K=32) -> f32
        gemm_tdm<2, true, false><<<dim3(DINNER / 64, TOKENS / 128), 256, 0, stream>>>(
            xdh, dtW_l, dtb_l, dtf, nullptr, DTRANK, 64, DTRANK, DINNER);

        // selective scan + D-skip + silu(z) gate -> y (f16)
        scan_kernel<<<dim3(DINNER / 256, BATCH), 256, 0, stream>>>(
            dtf, ucf, xdf, xz, Alog_l, D_l, yh);

        // x = y @ out_W^T   (M=8192, N=512, K=1024) -> f16 only
        gemm_tdm<0, false, true><<<dim3(DMODEL / 64, TOKENS / 128), 256, 0, stream>>>(
            yh, outW_l, nullptr, nullptr, xh, DINNER, DINNER, DINNER, DMODEL);
    }

    // out = x @ W_out^T + b_out   (M=8192, N=256, K=512) -> f32
    gemm_tdm<1, true, false><<<dim3(OUTDIM / 64, TOKENS / 128), 256, 0, stream>>>(
        xh, WoutH, b_out, out, nullptr, DMODEL, DMODEL, DMODEL, OUTDIM);
}
